// NELayer_146028888089
// MI455X (gfx1250) — matrix operand order repaired and checked by reading, not compiled, not run on hardware
//
#include <hip/hip_runtime.h>

typedef __attribute__((ext_vector_type(4)))  _Float16 v4h;
typedef __attribute__((ext_vector_type(8)))  _Float16 v8h;
typedef __attribute__((ext_vector_type(16))) _Float16 v16h;
typedef __attribute__((ext_vector_type(8)))  float    v8f;

#define N_NODES  50000
#define N_EDGES  800000
#define IN_NF    64
#define EDGES_NF 16
#define H_NF     128
#define OUT_NF   64

#define KE      160   // edge GEMM K padded (2*64+16 = 144 -> 160)
#define KN1     192   // node GEMM1 K (64 + 128)
#define KN2     128   // node GEMM2 K

#define KE_PAD  168   // LDS row stride (halves), 336B -> 16B aligned
#define KN1_PAD 200   // 400B row stride
#define H_PAD   136   // 272B row stride

__device__ __forceinline__ void cvt4(const float4 f, _Float16* p) {
    v4h h;
    h[0] = (_Float16)f.x; h[1] = (_Float16)f.y;
    h[2] = (_Float16)f.z; h[3] = (_Float16)f.w;
    *(v4h*)p = h;
}

__device__ __forceinline__ v8f wmma16(v16h a, v16h b, v8f c) {
    return __builtin_amdgcn_wmma_f32_16x16x32_f16(false, a, false, b,
                                                  (short)0, c, false, false);
}

// ---------------------------------------------------------------------------
// Prep: zero agg scratch
__global__ void zero_f32(float* __restrict__ p, int n) {
    int i = blockIdx.x * blockDim.x + threadIdx.x;
    if (i < n) p[i] = 0.0f;
}

// Prep: W[Ksrc][Ncols] f32 (row-major) -> WT[Ncols][Kpad] f16, zero-padded K
__global__ void cvt_weight(const float* __restrict__ src, _Float16* __restrict__ dst,
                           int Ksrc, int Kpad, int Ncols) {
    int idx = blockIdx.x * blockDim.x + threadIdx.x;
    if (idx >= Ncols * Kpad) return;
    int n = idx / Kpad;
    int k = idx - n * Kpad;
    dst[idx] = (k < Ksrc) ? (_Float16)src[(size_t)k * Ncols + n] : (_Float16)0.0f;
}

// ---------------------------------------------------------------------------
// Edge model: gather -> concat -> GEMM(144->128) -> ReLU -> store + scatter-add
__global__ __launch_bounds__(256) void edge_kernel(
    const float* __restrict__ node_feats,
    const int*   __restrict__ edge_index,    // [2][E]
    const float* __restrict__ edge_attr,     // [E][16]
    const _Float16* __restrict__ wte,        // [128][KE]  (W_e^T, f16, padded)
    const float* __restrict__ b_e,
    float* __restrict__ edge_feats,          // [E][128]
    float* __restrict__ agg)                 // [N][128], pre-zeroed
{
    __shared__ _Float16 sA[128 * KE_PAD];
    __shared__ unsigned sIdx[128];

    const int tid   = threadIdx.x;
    const int ebase = blockIdx.x * 128;

    // ---- stage A tile: [src(64) | tgt(64) | attr(16) | 0(16)] as f16 ----
    {
        const int row  = tid >> 1;
        const int half = tid & 1;
        const int e = ebase + row;
        const int i = edge_index[e];
        const int j = edge_index[N_EDGES + e];
        if (half == 0) sIdx[row] = (unsigned)i;
        _Float16* dst = &sA[row * KE_PAD];
        const float* src = node_feats + (size_t)(unsigned)i * IN_NF;
        const float* tgt = node_feats + (size_t)(unsigned)j * IN_NF;
        if (half == 0) {
            #pragma unroll
            for (int c = 0; c < 64; c += 4)
                cvt4(*(const float4*)(src + c), dst + c);
            #pragma unroll
            for (int c = 0; c < 8; c += 4)
                cvt4(*(const float4*)(tgt + c), dst + 64 + c);
        } else {
            #pragma unroll
            for (int c = 8; c < 64; c += 4)
                cvt4(*(const float4*)(tgt + c), dst + 64 + c);
            const float* ea = edge_attr + (size_t)(unsigned)e * EDGES_NF;
            #pragma unroll
            for (int c = 0; c < 16; c += 4)
                cvt4(*(const float4*)(ea + c), dst + 128 + c);
            v4h z = {(_Float16)0, (_Float16)0, (_Float16)0, (_Float16)0};
            #pragma unroll
            for (int c = 144; c < 160; c += 4)
                *(v4h*)(dst + c) = z;
        }
    }
    __syncthreads();

    const int wave  = tid >> 5;
    const int lane  = tid & 31;
    const int nloc  = lane & 15;
    const int khalf = (lane >> 4) << 4;   // B-matrix: upper lanes hold K=16..31
    const int koff  = (lane >> 4) << 3;   // C/D: upper lanes hold rows 8..15
    const int col   = wave * 16 + nloc;   // this lane's output column

    // B fragments, resident in VGPRs across all M tiles
    v16h bfrag[5];
    #pragma unroll
    for (int c = 0; c < 5; ++c)
        bfrag[c] = *(const v16h*)(wte + (size_t)col * KE + c * 32 + khalf);

    const float bias = b_e[col];
    const unsigned colu = (unsigned)col;

    #pragma unroll
    for (int m = 0; m < 8; ++m) {
        const _Float16* ap = &sA[(m * 16 + nloc) * KE_PAD];
        v8f acc{};
        #pragma unroll
        for (int c = 0; c < 5; ++c) {
            v8h lo = *(const v8h*)(ap + c * 32 + khalf);
            v8h hi = *(const v8h*)(ap + c * 32 + khalf + 8);
            v16h a = __builtin_shufflevector(lo, hi,
                     0,1,2,3,4,5,6,7,8,9,10,11,12,13,14,15);
            acc = wmma16(a, bfrag[c], acc);
        }
        const int rbase = m * 16 + koff;   // C row = r + (lane>=16 ? 8 : 0)
        // unsigned 32-bit offsets -> SADDR + zext(VOFFSET) addressing
        const unsigned ebank = (unsigned)(ebase + rbase) * H_NF + colu;
        #pragma unroll
        for (int r = 0; r < 8; ++r) {
            float v = acc[r] + bias;
            v = v > 0.0f ? v : 0.0f;
            edge_feats[(size_t)(ebank + (unsigned)r * H_NF)] = v;
            const unsigned aoff = sIdx[rbase + r] * H_NF + colu;
            atomicAdd(&agg[(size_t)aoff], v);
        }
    }
}

// ---------------------------------------------------------------------------
// Node model: concat(node,agg) -> GEMM(192->128) -> ReLU -> GEMM(128->64)
__global__ __launch_bounds__(256) void node_kernel(
    const float* __restrict__ node_feats,
    const float* __restrict__ agg,
    const _Float16* __restrict__ wtn1,   // [128][KN1]
    const float* __restrict__ b_n1,
    const _Float16* __restrict__ wtn2,   // [64][KN2]
    const float* __restrict__ b_n2,
    float* __restrict__ node_out)        // [N][64]
{
    __shared__ _Float16 sA[128 * KN1_PAD];
    __shared__ _Float16 sH[128 * H_PAD];

    const int tid   = threadIdx.x;
    const int nbase = blockIdx.x * 128;

    // ---- stage A tile: [node_feats(64) | agg(128)] as f16 ----
    {
        const int row  = tid >> 1;
        const int half = tid & 1;
        const int node = nbase + row;
        _Float16* dst = &sA[row * KN1_PAD];
        if (node < N_NODES) {
            const float* nf = node_feats + (size_t)(unsigned)node * IN_NF;
            const float* ag = agg + (size_t)(unsigned)node * H_NF;
            if (half == 0) {
                #pragma unroll
                for (int c = 0; c < 64; c += 4)
                    cvt4(*(const float4*)(nf + c), dst + c);
                #pragma unroll
                for (int c = 0; c < 32; c += 4)
                    cvt4(*(const float4*)(ag + c), dst + 64 + c);
            } else {
                #pragma unroll
                for (int c = 32; c < 128; c += 4)
                    cvt4(*(const float4*)(ag + c), dst + 64 + c);
            }
        } else {
            v4h z = {(_Float16)0, (_Float16)0, (_Float16)0, (_Float16)0};
            const int c0 = (half == 0) ? 0 : 96;
            #pragma unroll
            for (int c = 0; c < 96; c += 4)
                *(v4h*)(dst + c0 + c) = z;
        }
    }
    __syncthreads();

    const int wave  = tid >> 5;
    const int lane  = tid & 31;
    const int nloc  = lane & 15;
    const int khalf = (lane >> 4) << 4;
    const int koff  = (lane >> 4) << 3;

    // ---- GEMM1: K=192, each wave owns one 16-col tile of the 128 outputs ----
    {
        const int col = wave * 16 + nloc;
        v16h b1[6];
        #pragma unroll
        for (int c = 0; c < 6; ++c)
            b1[c] = *(const v16h*)(wtn1 + (size_t)col * KN1 + c * 32 + khalf);
        const float bias1 = b_n1[col];

        #pragma unroll
        for (int m = 0; m < 8; ++m) {
            const _Float16* ap = &sA[(m * 16 + nloc) * KN1_PAD];
            v8f acc{};
            #pragma unroll
            for (int c = 0; c < 6; ++c) {
                v8h lo = *(const v8h*)(ap + c * 32 + khalf);
                v8h hi = *(const v8h*)(ap + c * 32 + khalf + 8);
                v16h a = __builtin_shufflevector(lo, hi,
                         0,1,2,3,4,5,6,7,8,9,10,11,12,13,14,15);
                acc = wmma16(a, b1[c], acc);
            }
            const int rbase = m * 16 + koff;
            #pragma unroll
            for (int r = 0; r < 8; ++r) {
                float v = acc[r] + bias1;
                v = v > 0.0f ? v : 0.0f;
                sH[(rbase + r) * H_PAD + col] = (_Float16)v;
            }
        }
    }
    __syncthreads();

    // ---- GEMM2: K=128, N=64 -> 4 col tiles; waves 0-3 do M0..3, 4-7 do M4..7
    {
        const int col2 = (wave & 3) * 16 + nloc;
        v16h b2[4];
        #pragma unroll
        for (int c = 0; c < 4; ++c)
            b2[c] = *(const v16h*)(wtn2 + (size_t)col2 * KN2 + c * 32 + khalf);
        const float bias2 = b_n2[col2];
        const int m0 = (wave >> 2) * 4;

        #pragma unroll
        for (int mm = 0; mm < 4; ++mm) {
            const int m = m0 + mm;
            const _Float16* ap = &sH[(m * 16 + nloc) * H_PAD];
            v8f acc{};
            #pragma unroll
            for (int c = 0; c < 4; ++c) {
                v8h lo = *(const v8h*)(ap + c * 32 + khalf);
                v8h hi = *(const v8h*)(ap + c * 32 + khalf + 8);
                v16h a = __builtin_shufflevector(lo, hi,
                         0,1,2,3,4,5,6,7,8,9,10,11,12,13,14,15);
                acc = wmma16(a, b2[c], acc);
            }
            const int rbase = m * 16 + koff;
            #pragma unroll
            for (int r = 0; r < 8; ++r) {
                const int node = nbase + rbase + r;
                if (node < N_NODES) {
                    const unsigned ooff = (unsigned)node * OUT_NF + (unsigned)col2;
                    node_out[(size_t)ooff] = acc[r] + bias2;
                }
            }
        }
    }
}

// ---------------------------------------------------------------------------
extern "C" void kernel_launch(void* const* d_in, const int* in_sizes, int n_in,
                              void* d_out, int out_size, void* d_ws, size_t ws_size,
                              hipStream_t stream) {
    const float* node_feats = (const float*)d_in[0];
    const int*   edge_index = (const int*)  d_in[1];
    const float* edge_attr  = (const float*)d_in[2];
    const float* W_e  = (const float*)d_in[3];
    const float* b_e  = (const float*)d_in[4];
    const float* W_n1 = (const float*)d_in[5];
    const float* b_n1 = (const float*)d_in[6];
    const float* W_n2 = (const float*)d_in[7];
    const float* b_n2 = (const float*)d_in[8];

    float* node_out   = (float*)d_out;                          // [N][64]
    float* edge_feats = node_out + (size_t)N_NODES * OUT_NF;    // [E][128]

    // workspace layout: agg f32 [N][128], then f16 transposed weights
    const size_t AGG_ELEMS = (size_t)N_NODES * H_NF;
    char* ws = (char*)d_ws;
    float*     agg  = (float*)ws;
    _Float16*  wte  = (_Float16*)(ws + AGG_ELEMS * sizeof(float));
    _Float16*  wtn1 = wte  + (size_t)H_NF * KE;
    _Float16*  wtn2 = wtn1 + (size_t)H_NF * KN1;

    // prep
    {
        int n = (int)AGG_ELEMS;
        zero_f32<<<(n + 255) / 256, 256, 0, stream>>>(agg, n);
    }
    cvt_weight<<<(H_NF * KE   + 255) / 256, 256, 0, stream>>>(W_e,  wte,  2*IN_NF+EDGES_NF, KE,  H_NF);
    cvt_weight<<<(H_NF * KN1  + 255) / 256, 256, 0, stream>>>(W_n1, wtn1, H_NF+IN_NF,       KN1, H_NF);
    cvt_weight<<<(OUT_NF * KN2 + 255) / 256, 256, 0, stream>>>(W_n2, wtn2, H_NF,            KN2, OUT_NF);

    // edge model + scatter-add
    edge_kernel<<<N_EDGES / 128, 256, 0, stream>>>(
        node_feats, edge_index, edge_attr, wte, b_e, edge_feats, agg);

    // node MLP
    node_kernel<<<(N_NODES + 127) / 128, 256, 0, stream>>>(
        node_feats, agg, wtn1, b_n1, wtn2, b_n2, node_out);
}
